// LSTMDecoder_38517266711127
// MI455X (gfx1250) — compile-verified, matrix operand order checked by reading, not run on hardware
//
#include <hip/hip_runtime.h>
#include <hip/hip_bf16.h>

// ---------------- problem constants ----------------
#define BB    32
#define TT    128
#define QQ    64
#define HH    1024
#define AAD   256
#define RULED 512
#define ENCD  1024
#define IND   1280
#define G4H   4096
#define KCAT  3072      // H + ENC + H  (fused recurrent K)
#define KSPLIT 8        // split-K factor for the recurrent GEMM
#define NEGV  -1000000000.0f

typedef __bf16 bf16;
typedef __attribute__((ext_vector_type(8)))  bf16  bf16x8;
typedef __attribute__((ext_vector_type(16))) bf16  bf16x16;
typedef __attribute__((ext_vector_type(8)))  float f32x8;

// ---------------- WMMA fragment load ----------------
// 16-bit A/B fragment layout (CDNA5 ISA 7.12.2):
// lane L (0..15):  row = L,     K = {0..7} in v0..3,  K = {16..23} in v4..7
// lane L (16..31): row = L-16,  K = {8..15} in v0..3, K = {24..31} in v4..7
__device__ __forceinline__ bf16x16 load_frag(const bf16* __restrict__ base, int ld,
                                             int r0, int k, int lane) {
    int row = r0 + (lane & 15);
    int kk  = k + ((lane >> 4) << 3);
    const bf16* p = base + (size_t)row * ld + kk;
    bf16x8 lo = *(const bf16x8*)(p);
    bf16x8 hi = *(const bf16x8*)(p + 16);
    return __builtin_shufflevector(lo, hi, 0,1,2,3,4,5,6,7,8,9,10,11,12,13,14,15);
}

// ---------------- generic WMMA GEMM (register-blocked, optional split-K) ----
// D = A[M,K](bf16,lda) * Bt[N,K](bf16,ldb = B col-major) (+Cin)(+bias)
// grid = (N/(16*NT), M/(16*MT), KS); block = 32 threads = 1 wave.
// Wave computes a (16*MT)x(16*NT) tile over K-chunk [z*KC, z*KC+KC).
// MT=2,NT=4 keeps ~120 live VGPRs: no spills, ~8 waves/SIMD for latency hiding.
// Split-K (KS>1): each z writes its own fp32 slab at D + z*zstride
// (deterministic; reduced later). Cin/bias must be null in that mode.
template<int MT, int NT>
__global__ void wmma_gemm_bf16(const bf16* __restrict__ A, const bf16* __restrict__ Bt,
                               const float* __restrict__ Cin, long long ldc_in,
                               const float* __restrict__ bias,
                               float* __restrict__ D, long long zstride,
                               int KC, int lda, int ldb, int ldd)
{
    const int lane = threadIdx.x;
    const int mb = blockIdx.y * (16 * MT);
    const int n0 = blockIdx.x * (16 * NT);
    const int k0 = blockIdx.z * KC;
    const int kend = k0 + KC;
    const int rr = (lane >> 4) << 3;   // C/D: vgpr r holds row r + 8*(lane>>4)
    const int cl = lane & 15;          // C/D: col = lane & 15
    float* __restrict__ Dz = D + (size_t)blockIdx.z * zstride;

    f32x8 acc[MT][NT];
#pragma unroll
    for (int i = 0; i < MT; ++i)
#pragma unroll
        for (int j = 0; j < NT; ++j) {
            if (Cin) {
#pragma unroll
                for (int r = 0; r < 8; ++r)
                    acc[i][j][r] = Cin[(long long)(mb + 16 * i + rr + r) * ldc_in
                                       + n0 + j * 16 + cl];
            } else {
#pragma unroll
                for (int r = 0; r < 8; ++r) acc[i][j][r] = 0.f;
            }
        }

    // Incrementally-advanced speculative prefetch pointers (no per-iteration
    // address re-derivation, no branch; GLOBAL_PREFETCH_B8 is non-faulting).
    // lane spans 32 rows: covers the 32-row A panel and both halves of the
    // 64-row B panel.
    const bf16* pfA  = A  + (size_t)(mb + lane) * lda + k0 + 64;
    const bf16* pfB0 = Bt + (size_t)(n0 + lane) * ldb + k0 + 64;
    const bf16* pfB1 = Bt + (size_t)(n0 + 32 + (lane & (16 * NT - 33))) * ldb + k0 + 64;

    for (int k = k0; k < kend; k += 32) {
        bf16x16 a[MT];
#pragma unroll
        for (int i = 0; i < MT; ++i)
            a[i] = load_frag(A, lda, mb + 16 * i, k, lane);

        __builtin_prefetch(pfA, 0, 3);  pfA  += 32;
        __builtin_prefetch(pfB0, 0, 3); pfB0 += 32;
        if (NT > 2) { __builtin_prefetch(pfB1, 0, 3); pfB1 += 32; }

#pragma unroll
        for (int j = 0; j < NT; ++j) {
            bf16x16 b = load_frag(Bt, ldb, n0 + j * 16, k, lane);
#pragma unroll
            for (int i = 0; i < MT; ++i)
                acc[i][j] = __builtin_amdgcn_wmma_f32_16x16x32_bf16(
                    false, a[i], false, b, (short)0, acc[i][j], false, false);
        }
    }

#pragma unroll
    for (int i = 0; i < MT; ++i)
#pragma unroll
        for (int j = 0; j < NT; ++j) {
            float bv = bias ? bias[n0 + j * 16 + cl] : 0.f;
#pragma unroll
            for (int r = 0; r < 8; ++r)
                Dz[(size_t)(mb + 16 * i + rr + r) * ldd + n0 + j * 16 + cl] =
                    acc[i][j][r] + bv;
        }
}

// ---------------- split-K reduce: g = XW[:,t,:] + sum_z P[z] ----------------
__global__ void reduce_g(const float* __restrict__ P, const float* __restrict__ XW,
                         float* __restrict__ g, int t) {
    int i = blockIdx.x * 256 + threadIdx.x;       // i in [0, BB*G4H)
    int b = i >> 12;                               // /4096
    int n = i & (G4H - 1);
    float s = XW[((size_t)b * TT + t) * G4H + n];
#pragma unroll
    for (int z = 0; z < KSPLIT; ++z)
        s += P[(size_t)z * BB * G4H + i];
    g[i] = s;
}

// ---------------- conversion kernels ----------------
__global__ void cvt_f32_bf16(const float* __restrict__ in, bf16* __restrict__ out, int n) {
    int i = blockIdx.x * 256 + threadIdx.x;
    if (i < n) out[i] = (bf16)in[i];
}

// in: [K,N] fp32 -> out[n*ldo + koff + k] bf16 (transpose into a column slice)
__global__ void transpose_to_bf16(const float* __restrict__ in, bf16* __restrict__ out,
                                  int K, int N, int ldo, int koff) {
    __shared__ float tile[32][33];
    int n0 = blockIdx.x * 32, k0 = blockIdx.y * 32;
    int tx = threadIdx.x, ty = threadIdx.y;
#pragma unroll
    for (int i = 0; i < 32; i += 8)
        tile[ty + i][tx] = in[(size_t)(k0 + ty + i) * N + n0 + tx];
    __syncthreads();
#pragma unroll
    for (int i = 0; i < 32; i += 8)
        out[(size_t)(n0 + ty + i) * ldo + koff + k0 + tx] = (bf16)tile[tx][ty + i];
}

// ---------------- attention kernels ----------------
// score[b,q] = sum_a tanh(ctx_att[b,q,a] + hW2[b,a]) * attV[a], masked.
__global__ void attn_score(const float* __restrict__ ctx_att, const float* __restrict__ hW2,
                           const float* __restrict__ attV, const float* __restrict__ cmask,
                           float* __restrict__ scores) {
    __shared__ float red[256];
    int q = blockIdx.x, b = blockIdx.y, a = threadIdx.x;
    float v = tanhf(ctx_att[((size_t)b * QQ + q) * AAD + a] + hW2[b * AAD + a]) * attV[a];
    red[a] = v;
    __syncthreads();
    for (int s = 128; s > 0; s >>= 1) {
        if (a < s) red[a] += red[a + s];
        __syncthreads();
    }
    if (a == 0)
        scores[b * QQ + q] = (cmask[b * QQ + q] > 0.f) ? red[0] : NEGV;
}

// softmax over Q, then ctx_vec[b,e]; writes Acat ctx slot + bf16 history.
__global__ void softmax_ctx(const float* __restrict__ scores, const float* __restrict__ context,
                            bf16* __restrict__ Acat, bf16* __restrict__ ctxs, int t) {
    __shared__ float al[QQ];
    int b = blockIdx.x, tid = threadIdx.x;
    if (tid < QQ) al[tid] = scores[b * QQ + tid];
    __syncthreads();
    if (tid == 0) {
        float mx = al[0];
        for (int q = 1; q < QQ; ++q) mx = fmaxf(mx, al[q]);
        float s = 0.f;
        for (int q = 0; q < QQ; ++q) { al[q] = expf(al[q] - mx); s += al[q]; }
        float inv = 1.f / s;
        for (int q = 0; q < QQ; ++q) al[q] *= inv;
    }
    __syncthreads();
    for (int e = tid; e < ENCD; e += 256) {
        float s = 0.f;
#pragma unroll 8
        for (int q = 0; q < QQ; ++q)
            s += al[q] * context[((size_t)b * QQ + q) * ENCD + e];
        bf16 v = (bf16)s;
        Acat[(size_t)b * KCAT + HH + e] = v;                 // ctx slot of fused A
        ctxs[((size_t)b * TT + t) * ENCD + e] = v;
    }
}

// par slot of fused A: Acat[b, 2048+e] = hist[b, par[b,t], e]
__global__ void gather_par(const int* __restrict__ par, const bf16* __restrict__ hist,
                           bf16* __restrict__ Acat, int t) {
    int b = blockIdx.x;
    int p = par[b * TT + t];
    for (int e = threadIdx.x; e < HH; e += 256)
        Acat[(size_t)b * KCAT + 2 * HH + e] = hist[((size_t)b * TT + p) * HH + e];
}

// ---------------- LSTM pointwise ----------------
__device__ __forceinline__ float sigm(float x) { return 1.f / (1.f + expf(-x)); }

__global__ void lstm_point(const float* __restrict__ g, const float* __restrict__ mask,
                           float* __restrict__ h, float* __restrict__ c,
                           bf16* __restrict__ Acat, bf16* __restrict__ hist_bf, int t) {
    int b = blockIdx.x;
    float m = mask[b * TT + t];
    const float* gb = g + (size_t)b * G4H;
    for (int i = threadIdx.x; i < HH; i += 256) {
        float ig = gb[i], fg = gb[HH + i], cg = gb[2 * HH + i], og = gb[3 * HH + i];
        float c_old = c[b * HH + i], h_old = h[b * HH + i];
        float cn = sigm(fg) * c_old + sigm(ig) * tanhf(cg);
        float hn = sigm(og) * tanhf(cn);
        float hu = m * hn + (1.f - m) * h_old;
        float cu = m * cn + (1.f - m) * c_old;
        h[b * HH + i] = hu;
        c[b * HH + i] = cu;
        bf16 hv = (bf16)hu;
        Acat[(size_t)b * KCAT + i] = hv;                      // h slot of fused A
        hist_bf[((size_t)b * TT + t) * HH + i] = hv;
    }
}

// ---------------- host orchestration ----------------
extern "C" void kernel_launch(void* const* d_in, const int* in_sizes, int n_in,
                              void* d_out, int out_size, void* d_ws, size_t ws_size,
                              hipStream_t stream) {
    (void)in_sizes; (void)n_in; (void)out_size;
    const float* X      = (const float*)d_in[0];   // [B,T,IN]
    const float* ctxIn  = (const float*)d_in[1];   // [B,Q,ENC]
    const float* cmask  = (const float*)d_in[2];   // [B,Q]
    const float* amask  = (const float*)d_in[3];   // [B,T]
    const int*   par    = (const int*)  d_in[4];   // [B,T]
    const float* W_x    = (const float*)d_in[5];
    const float* b_x    = (const float*)d_in[6];
    const float* U_h    = (const float*)d_in[7];
    const float* C_ctx  = (const float*)d_in[8];
    const float* H_par  = (const float*)d_in[9];
    const float* attW1  = (const float*)d_in[10];
    const float* attb1  = (const float*)d_in[11];
    const float* attW2  = (const float*)d_in[12];
    const float* attV   = (const float*)d_in[13];
    const float* W_rule = (const float*)d_in[14];
    const float* b_rule = (const float*)d_in[15];
    const float* W_tok  = (const float*)d_in[16];
    const float* b_tok  = (const float*)d_in[17];

    // ---- workspace carve (256B-aligned bump allocator) ----
    char* w = (char*)d_ws;
    size_t used = 0;
    auto alloc = [&](size_t bytes) -> void* {
        void* p = w + used;
        used += (bytes + 255) & ~(size_t)255;
        return p;
    };
    float* XW      = (float*)alloc((size_t)BB * TT * G4H * 4);    // 64 MB
    float* ctx_att = (float*)alloc((size_t)BB * QQ * AAD * 4);
    float* hW2     = (float*)alloc((size_t)BB * AAD * 4);
    float* scores  = (float*)alloc((size_t)BB * QQ * 4);
    float* g       = (float*)alloc((size_t)BB * G4H * 4);
    float* Pbuf    = (float*)alloc((size_t)KSPLIT * BB * G4H * 4); // split-K partials
    float* h       = (float*)alloc((size_t)BB * HH * 4);
    float* c       = (float*)alloc((size_t)BB * HH * 4);
    bf16*  Acat    = (bf16*) alloc((size_t)BB * KCAT * 2);        // [h | ctx | par]
    bf16*  hist_bf = (bf16*) alloc((size_t)BB * TT * HH * 2);     // hidden history (bf16)
    bf16*  ctxs_bf = (bf16*) alloc((size_t)BB * TT * ENCD * 2);   // ctx history (bf16)
    bf16*  X_bf    = (bf16*) alloc((size_t)BB * TT * IND * 2);
    bf16*  ctxE_bf = (bf16*) alloc((size_t)BB * QQ * ENCD * 2);
    bf16*  Wx_t    = (bf16*) alloc((size_t)G4H * IND * 2);        // [4H, IN]
    bf16*  Bcat    = (bf16*) alloc((size_t)G4H * KCAT * 2);       // [4H, H|ENC|H]
    bf16*  W1_t    = (bf16*) alloc((size_t)AAD * ENCD * 2);
    bf16*  W2_t    = (bf16*) alloc((size_t)AAD * HH * 2);
    bf16*  Wr_t    = (bf16*) alloc((size_t)RULED * HH * 2);
    bf16*  Wt_t    = (bf16*) alloc((size_t)RULED * (HH + ENCD) * 2);
    if (used > ws_size) return;  // workspace too small: bail deterministically

    // ---- zero-init recurrent state (ws is poisoned) ----
    hipMemsetAsync(h,       0, (size_t)BB * HH * 4, stream);
    hipMemsetAsync(c,       0, (size_t)BB * HH * 4, stream);
    hipMemsetAsync(Acat,    0, (size_t)BB * KCAT * 2, stream);
    hipMemsetAsync(hist_bf, 0, (size_t)BB * TT * HH * 2, stream);

    // ---- one-time conversions (captured in graph; deterministic) ----
    cvt_f32_bf16<<<(BB * TT * IND + 255) / 256, 256, 0, stream>>>(X, X_bf, BB * TT * IND);
    cvt_f32_bf16<<<(BB * QQ * ENCD + 255) / 256, 256, 0, stream>>>(ctxIn, ctxE_bf, BB * QQ * ENCD);

    dim3 tb(32, 8);
    transpose_to_bf16<<<dim3(G4H / 32, IND / 32),  tb, 0, stream>>>(W_x,   Wx_t, IND,  G4H, IND, 0);
    // fused recurrent weights: Bcat[n, 0:H]=U_h, [H:H+ENC]=C_ctx, [H+ENC:3H]=H_par
    transpose_to_bf16<<<dim3(G4H / 32, HH / 32),   tb, 0, stream>>>(U_h,   Bcat, HH,   G4H, KCAT, 0);
    transpose_to_bf16<<<dim3(G4H / 32, ENCD / 32), tb, 0, stream>>>(C_ctx, Bcat, ENCD, G4H, KCAT, HH);
    transpose_to_bf16<<<dim3(G4H / 32, HH / 32),   tb, 0, stream>>>(H_par, Bcat, HH,   G4H, KCAT, 2 * HH);
    transpose_to_bf16<<<dim3(AAD / 32, ENCD / 32), tb, 0, stream>>>(attW1, W1_t, ENCD, AAD, ENCD, 0);
    transpose_to_bf16<<<dim3(AAD / 32, HH / 32),   tb, 0, stream>>>(attW2, W2_t, HH,   AAD, HH, 0);
    transpose_to_bf16<<<dim3(RULED / 32, HH / 32), tb, 0, stream>>>(W_rule, Wr_t, HH,  RULED, HH, 0);
    transpose_to_bf16<<<dim3(RULED / 32, (HH + ENCD) / 32), tb, 0, stream>>>(
        W_tok, Wt_t, HH + ENCD, RULED, HH + ENCD, 0);

    // ---- parallel pre-GEMMs ----
    // XW = X @ W_x + b_x : [4096,1280]x[1280,4096]
    wmma_gemm_bf16<2, 4><<<dim3(G4H / 64, (BB * TT) / 32, 1), 32, 0, stream>>>(
        X_bf, Wx_t, nullptr, 0, b_x, XW, 0, IND, IND, IND, G4H);
    // ctx_att = context @ att_W1 + att_b1 : [2048,1024]x[1024,256]
    wmma_gemm_bf16<2, 4><<<dim3(AAD / 64, (BB * QQ) / 32, 1), 32, 0, stream>>>(
        ctxE_bf, W1_t, nullptr, 0, attb1, ctx_att, 0, ENCD, ENCD, ENCD, AAD);

    // ---- sequential scan over T ----
    const int KC = KCAT / KSPLIT;  // 384
    for (int t = 0; t < TT; ++t) {
        // hW2 = h @ att_W2 (reads h slot = first H columns of Acat)
        wmma_gemm_bf16<2, 4><<<dim3(AAD / 64, BB / 32, 1), 32, 0, stream>>>(
            Acat, W2_t, nullptr, 0, nullptr, hW2, 0, HH, KCAT, HH, AAD);
        attn_score<<<dim3(QQ, BB), 256, 0, stream>>>(ctx_att, hW2, attV, cmask, scores);
        softmax_ctx<<<BB, 256, 0, stream>>>(scores, ctxIn, Acat, ctxs_bf, t);
        gather_par<<<BB, 256, 0, stream>>>(par, hist_bf, Acat, t);

        // fused recurrent GEMM (split-K partials): [32,3072] x [3072,4096]
        wmma_gemm_bf16<2, 4><<<dim3(G4H / 64, BB / 32, KSPLIT), 32, 0, stream>>>(
            Acat, Bcat, nullptr, 0, nullptr, Pbuf, (long long)BB * G4H,
            KC, KCAT, KCAT, G4H);
        // g = XW[:,t,:] + sum_z partials (fixed order -> deterministic)
        reduce_g<<<(BB * G4H) / 256, 256, 0, stream>>>(Pbuf, XW, g, t);

        lstm_point<<<BB, 256, 0, stream>>>(g, amask, h, c, Acat, hist_bf, t);
    }

    // ---- output heads ----
    float* out_rule = (float*)d_out;                       // [B,T,512]
    float* out_tok  = out_rule + (size_t)BB * TT * RULED;  // [B,T,512]
    // trans_rule = hidden @ W_rule + b_rule
    wmma_gemm_bf16<2, 4><<<dim3(RULED / 64, (BB * TT) / 32, 1), 32, 0, stream>>>(
        hist_bf, Wr_t, nullptr, 0, b_rule, out_rule, 0, HH, HH, HH, RULED);
    // trans_token = [hidden|ctx] @ W_tok + b_tok  (top rows on hidden, bottom on ctx)
    wmma_gemm_bf16<2, 4><<<dim3(RULED / 64, (BB * TT) / 32, 1), 32, 0, stream>>>(
        hist_bf, Wt_t, nullptr, 0, b_tok, out_tok, 0, HH, HH, HH + ENCD, RULED);
    wmma_gemm_bf16<2, 4><<<dim3(RULED / 64, (BB * TT) / 32, 1), 32, 0, stream>>>(
        ctxs_bf, Wt_t + HH, out_tok, RULED, nullptr, out_tok, 0, ENCD, ENCD, HH + ENCD, RULED);
}
